// LinearAttention_84559316123809
// MI455X (gfx1250) — matrix operand order repaired and checked by reading, not compiled, not run on hardware
//
#include <hip/hip_runtime.h>
#include <hip/hip_bf16.h>

// ---------------------------------------------------------------------------
// Linear attention for MI455X (gfx1250, wave32, WMMA + async LDS staging).
// Pipeline: cvt -> qkv GEMM (bf16 WMMA) -> softmax(k) -> context^T = v@k^T
//           (split-K WMMA + reduce) -> out = ctx^T @ q -> final GEMM + bias.
// Contiguous LDS staging uses GLOBAL_LOAD_ASYNC_TO_LDS_B128 (ASYNCcnt);
// B operand of NN GEMM is staged transposed + K-pair packed via VGPRs so
// fragment builds are ds_load_b128.
// ---------------------------------------------------------------------------

#define USE_ASYNC_LDS 1

typedef unsigned short ushort_t;
typedef __attribute__((ext_vector_type(16))) __bf16 v16bf;
typedef __attribute__((ext_vector_type(8)))  float  v8f;

union Frag16 {
    v16bf        v;
    unsigned int d[8];
    ushort_t     u[16];
};

__device__ __forceinline__ ushort_t f2bf(float f) {
    unsigned int u = __float_as_uint(f);
    u += 0x7FFFu + ((u >> 16) & 1u);     // round-to-nearest-even
    return (ushort_t)(u >> 16);
}
__device__ __forceinline__ float bf2f(ushort_t h) {
    return __uint_as_float(((unsigned int)h) << 16);
}

#if USE_ASYNC_LDS
// Copy 32 contiguous bytes global -> LDS via async DMA (2 x B128).
// IOFFSET is added to both the LDS and global addresses (ISA 10.7/15.18).
__device__ __forceinline__ void async_copy32(const void* lds_dst, const void* gsrc) {
    unsigned           ldsOff = (unsigned)(unsigned long long)lds_dst;   // ADDR[31:0] = LDS byte addr
    unsigned long long gaddr  = (unsigned long long)gsrc;
    asm volatile("global_load_async_to_lds_b128 %0, %1, off\n\t"
                 "global_load_async_to_lds_b128 %0, %1, off offset:16"
                 :: "v"(ldsOff), "v"(gaddr) : "memory");
}
__device__ __forceinline__ void async_wait0() {
    asm volatile("s_wait_asynccnt 0" ::: "memory");
}
#endif

// ------------------------------ convert ------------------------------------
__global__ void cvt_f32_bf16(const float* __restrict__ s, ushort_t* __restrict__ d, int n) {
    int i = blockIdx.x * blockDim.x + threadIdx.x;
    int stride = gridDim.x * blockDim.x;
    for (; i < n; i += stride) d[i] = f2bf(s[i]);
}

// --------------------- generic NN GEMM: C = A(MxK) @ B(KxN) ----------------
// A,B row-major bf16; C row-major (bf16 or f32 + optional bias).
// block = 128 threads (4 waves); 64x64 C tile; K staged in chunks of 32.
// LDS B layout: lds_bt[n*40 + kk] = B[k0+kk][tileN0+n], kk-pairs dword-packed.
#define BT_STRIDE 40   // ushorts per n-row (32 + 8 pad; 80B = 16B multiple)

__global__ __launch_bounds__(128)
void gemm_nn_wmma(const ushort_t* __restrict__ A, const ushort_t* __restrict__ B,
                  void* __restrict__ C, const float* __restrict__ bias,
                  int M, int N, int K, int innerCount,
                  long long aStrideZ, long long bOuter, long long bInner,
                  long long cOuter, long long cInner, int outF32)
{
    __shared__ __align__(16) ushort_t lds_a [64 * 32];
    __shared__ __align__(16) ushort_t lds_bt[64 * BT_STRIDE];

    const int z  = blockIdx.z;
    const int zo = z / innerCount;
    const int zi = z - zo * innerCount;
    const ushort_t* Ab = A + (long long)z * aStrideZ;
    const ushort_t* Bb = B + (long long)zo * bOuter + (long long)zi * bInner;
    const long long cOff = (long long)zo * cOuter + (long long)zi * cInner;

    const int tileM0 = blockIdx.y * 64;
    const int tileN0 = blockIdx.x * 64;
    const int tid    = threadIdx.x;
    const int wave   = tid >> 5;
    const int lane   = tid & 31;
    const int lane16 = lane & 15;
    const int hi     = lane >> 4;        // 0: lanes 0-15, 1: lanes 16-31

    v8f acc[4] = {};

    for (int k0 = 0; k0 < K; k0 += 32) {
        // stage A chunk 64x32 row-major (32B per thread)
        {
            int row = tid >> 1, half = tid & 1;
            const ushort_t* src = Ab + (size_t)(tileM0 + row) * K + k0 + half * 16;
#if USE_ASYNC_LDS
            async_copy32(&lds_a[row * 32 + half * 16], src);
#else
            uint4* dst = (uint4*)&lds_a[row * 32 + half * 16];
            dst[0] = ((const uint4*)src)[0]; dst[1] = ((const uint4*)src)[1];
#endif
        }
        // stage B chunk 32x64, transposed + K-pair packed into lds_bt
        {
            int kp = tid & 15;           // k-pair 0..15 (k = 2*kp, 2*kp+1)
            int n0 = (tid >> 4) * 8;     // 8 columns per thread
            const ushort_t* r0 = Bb + (size_t)(k0 + 2 * kp) * N + tileN0 + n0;
            uint4 lo4 = *(const uint4*)r0;
            uint4 hi4 = *(const uint4*)(r0 + N);
            const ushort_t* lo = (const ushort_t*)&lo4;
            const ushort_t* hp = (const ushort_t*)&hi4;
            #pragma unroll
            for (int i = 0; i < 8; ++i) {
                unsigned int dw = (unsigned int)lo[i] | ((unsigned int)hp[i] << 16);
                *(unsigned int*)&lds_bt[(n0 + i) * BT_STRIDE + 2 * kp] = dw;
            }
        }
        // prefetch next B chunk while we do LDS + WMMA work
        if (k0 + 32 < K) {
            __builtin_prefetch(Bb + (size_t)(k0 + 32 + lane) * N + tileN0, 0, 1);
        }
#if USE_ASYNC_LDS
        async_wait0();
#endif
        __syncthreads();

        // A fragment: 16x32, this wave's rows [wave*16, wave*16+16)
        Frag16 af;
        const int mrow = wave * 16 + lane16;
        #pragma unroll
        for (int g = 0; g < 8; ++g) {
            int kbase = 2 * g;
            int kk = kbase + hi * 16;                // contiguous k = 16*hi + s (matches B)
            af.d[g] = *(const unsigned int*)&lds_a[mrow * 32 + kk];
        }

        #pragma unroll
        for (int c = 0; c < 4; ++c) {
            Frag16 bfrag;
            const int ncol = c * 16 + lane16;
            #pragma unroll
            for (int j = 0; j < 8; ++j)              // contiguous -> 2x ds_load_b128
                bfrag.d[j] = *(const unsigned int*)&lds_bt[ncol * BT_STRIDE + hi * 16 + 2 * j];
            acc[c] = __builtin_amdgcn_wmma_f32_16x16x32_bf16(
                false, af.v, false, bfrag.v, (short)0, acc[c], false, false);
        }
        __syncthreads();
    }

    // C/D layout: VGPR r -> M = r (lanes 0-15) / 8+r (lanes 16-31), N = lane16
    #pragma unroll
    for (int c = 0; c < 4; ++c) {
        #pragma unroll
        for (int r = 0; r < 8; ++r) {
            int row = tileM0 + wave * 16 + hi * 8 + r;
            int col = tileN0 + c * 16 + lane16;
            float val = acc[c][r];
            if (bias) val += bias[row];
            size_t idx = (size_t)(cOff + (long long)row * N + col);
            if (outF32) ((float*)C)[idx] = val;
            else        ((ushort_t*)C)[idx] = f2bf(val);
        }
    }
}

// ---------------- NT GEMM: ctx^T[e,d] = sum_n v[e,n] * k[d,n] --------------
// Split-K: blockIdx.x = K-chunk of 512 (8 chunks of L=4096); fp32 partials.
// Both operands row-major along L, so B^T K-pairs are naturally contiguous,
// and both stages are straight copies -> async load-to-LDS.
__global__ __launch_bounds__(128)
void gemm_nt_wmma(const ushort_t* __restrict__ qkv, float* __restrict__ ctx_part, int L)
{
    __shared__ __align__(16) ushort_t lds_a [64 * 32];   // v rows
    __shared__ __align__(16) ushort_t lds_bt[64 * 32];   // k rows (B^T layout)

    const int part = blockIdx.x;        // 0..7
    const int z    = blockIdx.z;        // b*8 + h
    const int b    = z >> 3, h = z & 7;
    const ushort_t* Av = qkv + ((size_t)b * 1536 + 1024 + h * 64) * L;  // v
    const ushort_t* Bk = qkv + ((size_t)b * 1536 + 512  + h * 64) * L;  // softmax(k)

    const int tid = threadIdx.x, wave = tid >> 5, lane = tid & 31;
    const int lane16 = lane & 15, hi = lane >> 4;
    const int kstart = part * 512;

    v8f acc[4] = {};
    for (int k0 = 0; k0 < 512; k0 += 32) {
        {
            int row = tid >> 1, half = tid & 1;
            const ushort_t* sA = Av + (size_t)row * L + kstart + k0 + half * 16;
            const ushort_t* sB = Bk + (size_t)row * L + kstart + k0 + half * 16;
#if USE_ASYNC_LDS
            async_copy32(&lds_a [row * 32 + half * 16], sA);
            async_copy32(&lds_bt[row * 32 + half * 16], sB);
#else
            uint4* dA = (uint4*)&lds_a [row * 32 + half * 16];
            uint4* dB = (uint4*)&lds_bt[row * 32 + half * 16];
            dA[0] = ((const uint4*)sA)[0]; dA[1] = ((const uint4*)sA)[1];
            dB[0] = ((const uint4*)sB)[0]; dB[1] = ((const uint4*)sB)[1];
#endif
        }
#if USE_ASYNC_LDS
        async_wait0();
#endif
        __syncthreads();

        Frag16 af;
        const int mrow = wave * 16 + lane16;
        #pragma unroll
        for (int g = 0; g < 8; ++g) {
            int kbase = 2 * g;
            af.d[g] = *(const unsigned int*)&lds_a[mrow * 32 + kbase + hi * 16];
        }
        #pragma unroll
        for (int c = 0; c < 4; ++c) {
            Frag16 bfrag;
            const int ncol = c * 16 + lane16;
            #pragma unroll
            for (int j = 0; j < 8; ++j) {
                int kk = hi * 16 + 2 * j;   // contiguous pair along k-row
                bfrag.d[j] = *(const unsigned int*)&lds_bt[ncol * 32 + kk];
            }
            acc[c] = __builtin_amdgcn_wmma_f32_16x16x32_bf16(
                false, af.v, false, bfrag.v, (short)0, acc[c], false, false);
        }
        __syncthreads();
    }

    float* dst = ctx_part + ((size_t)z * 8 + part) * 4096;
    #pragma unroll
    for (int c = 0; c < 4; ++c) {
        #pragma unroll
        for (int r = 0; r < 8; ++r) {
            int row = wave * 16 + hi * 8 + r;
            int col = c * 16 + lane16;
            dst[row * 64 + col] = acc[c][r];
        }
    }
}

__global__ void reduce_ctx(const float* __restrict__ part, ushort_t* __restrict__ ctxT, int total) {
    int idx = blockIdx.x * 256 + threadIdx.x;
    if (idx >= total) return;
    int z = idx >> 12;          // / 4096
    int r = idx & 4095;
    float s = 0.f;
    #pragma unroll
    for (int p = 0; p < 8; ++p) s += part[((size_t)z * 8 + p) * 4096 + r];
    ctxT[idx] = f2bf(s);
}

// ---------------------- softmax over L on k rows (in place) ----------------
__global__ __launch_bounds__(256)
void softmax_k_rows(ushort_t* __restrict__ qkv, int L)
{
    __shared__ float red[256];
    const int blk = blockIdx.x;
    const int b = blk >> 9, i = blk & 511;
    ushort_t* row = qkv + ((size_t)b * 1536 + 512 + i) * L;
    const int tid = threadIdx.x;

    float v[16];
    float m = -1e30f;
    #pragma unroll
    for (int j = 0; j < 16; ++j) {
        float f = bf2f(row[j * 256 + tid]);
        v[j] = f;
        m = fmaxf(m, f);
    }
    red[tid] = m; __syncthreads();
    for (int s = 128; s > 0; s >>= 1) { if (tid < s) red[tid] = fmaxf(red[tid], red[tid + s]); __syncthreads(); }
    m = red[0]; __syncthreads();

    float sum = 0.f;
    #pragma unroll
    for (int j = 0; j < 16; ++j) { v[j] = __expf(v[j] - m); sum += v[j]; }
    red[tid] = sum; __syncthreads();
    for (int s = 128; s > 0; s >>= 1) { if (tid < s) red[tid] += red[tid + s]; __syncthreads(); }
    float inv = 1.0f / red[0];

    #pragma unroll
    for (int j = 0; j < 16; ++j) row[j * 256 + tid] = f2bf(v[j] * inv);
}

// ---------------------------------------------------------------------------
extern "C" void kernel_launch(void* const* d_in, const int* in_sizes, int n_in,
                              void* d_out, int out_size, void* d_ws, size_t ws_size,
                              hipStream_t stream)
{
    (void)in_sizes; (void)n_in; (void)out_size; (void)ws_size;
    const int B = 8, Cdim = 512, L = 4096, O3 = 1536;

    const float* x     = (const float*)d_in[0];   // (8,512,4096)
    const float* w_qkv = (const float*)d_in[1];   // (1536,512)
    const float* w_out = (const float*)d_in[2];   // (512,512)
    const float* b_out = (const float*)d_in[3];   // (512,)

    size_t off = 0;
    char* wsb = (char*)d_ws;
    auto wsAlloc = [&](size_t bytes) -> void* {
        void* p = wsb + off;
        off = (off + bytes + 255) & ~(size_t)255;
        return p;
    };
    ushort_t* x_bf     = (ushort_t*)wsAlloc((size_t)B * Cdim * L * 2);   // 32 MB
    ushort_t* wqkv_bf  = (ushort_t*)wsAlloc((size_t)O3 * Cdim * 2);      // 1.5 MB
    ushort_t* wout_bf  = (ushort_t*)wsAlloc((size_t)Cdim * Cdim * 2);    // 0.5 MB
    ushort_t* qkv_bf   = (ushort_t*)wsAlloc((size_t)B * O3 * L * 2);     // 96 MB
    float*    ctx_part = (float*)   wsAlloc((size_t)64 * 8 * 4096 * 4);  // 8 MB
    ushort_t* ctxT_bf  = (ushort_t*)wsAlloc((size_t)64 * 4096 * 2);      // 0.5 MB
    ushort_t* out_bf   = (ushort_t*)wsAlloc((size_t)B * Cdim * L * 2);   // 32 MB

    // 1) bf16 conversions
    cvt_f32_bf16<<<4096, 256, 0, stream>>>(x, x_bf, B * Cdim * L);
    cvt_f32_bf16<<<1024, 256, 0, stream>>>(w_qkv, wqkv_bf, O3 * Cdim);
    cvt_f32_bf16<<< 512, 256, 0, stream>>>(w_out, wout_bf, Cdim * Cdim);

    // 2) qkv = w_qkv @ x  (per batch):  M=1536, N=4096, K=512 -> bf16
    gemm_nn_wmma<<<dim3(64, 24, 8), 128, 0, stream>>>(
        wqkv_bf, x_bf, (void*)qkv_bf, nullptr,
        O3, L, Cdim, /*inner*/1,
        /*aStrideZ*/0LL, /*bOuter*/(long long)Cdim * L, /*bInner*/0LL,
        /*cOuter*/(long long)O3 * L, /*cInner*/0LL, /*outF32*/0);

    // 3) softmax over L on k rows (in place on qkv_bf)
    softmax_k_rows<<<B * 512, 256, 0, stream>>>(qkv_bf, L);

    // 4) ctx^T = v @ k^T per (b,h), split-K into 8 fp32 partials, then reduce
    gemm_nt_wmma<<<dim3(8, 1, 64), 128, 0, stream>>>(qkv_bf, ctx_part, L);
    reduce_ctx<<<1024, 256, 0, stream>>>(ctx_part, ctxT_bf, 64 * 4096);

    // 5) out = ctx^T @ q per (b,h): M=64, N=4096, K=64 -> bf16
    gemm_nn_wmma<<<dim3(64, 1, 64), 128, 0, stream>>>(
        ctxT_bf, qkv_bf, (void*)out_bf, nullptr,
        64, L, 64, /*inner*/8,
        /*aStrideZ*/4096LL, /*bOuter*/(long long)O3 * L, /*bInner*/64LL * L,
        /*cOuter*/(long long)Cdim * L, /*cInner*/64LL * L, /*outF32*/0);

    // 6) final = w_out @ out + bias (per batch): M=512, N=4096, K=512 -> f32
    gemm_nn_wmma<<<dim3(64, 8, 8), 128, 0, stream>>>(
        wout_bf, out_bf, d_out, b_out,
        Cdim, L, Cdim, /*inner*/1,
        /*aStrideZ*/0LL, /*bOuter*/(long long)Cdim * L, /*bInner*/0LL,
        /*cOuter*/(long long)Cdim * L, /*cInner*/0LL, /*outF32*/1);
}
